// SimplerPhxLearner_10488310136924
// MI455X (gfx1250) — matrix operand order repaired and checked by reading, not compiled, not run on hardware
//
#include <hip/hip_runtime.h>
#include <math.h>

// ---------------------------------------------------------------------------
// Problem dims
// ---------------------------------------------------------------------------
#define Bsz   16
#define Lsz   1024
#define D0sz  64
#define Hsz   512
#define G4    2048          // 4*H gate dim
#define KK    576           // D0 + H (concatenated [x|h] K dimension)
#define NWG   16            // persistent workgroups for recurrent kernels

// ---------------------------------------------------------------------------
// Workspace layout (bytes)
// ---------------------------------------------------------------------------
constexpr size_t SZ_BIGW  = (size_t)G4 * KK * 2;        // 2,359,296  bf16 [2048][576]
constexpr size_t SZ_SQW   = (size_t)Hsz * Hsz * 2;      // 524,288    bf16 [512][512]
constexpr size_t SZ_W3    = (size_t)D0sz * Hsz * 2;     // 65,536     bf16 [64][512]
constexpr size_t SZ_STATE = (size_t)Bsz * Hsz;          // 8192 elems
constexpr size_t SZ_SEQ   = (size_t)Bsz * Lsz * Hsz * 2;// 16,777,216 bf16 [B][L][H]

constexpr size_t O_ENCW = 0;
constexpr size_t O_DECW = O_ENCW + SZ_BIGW;
constexpr size_t O_W2   = O_DECW + SZ_BIGW;
constexpr size_t O_WQ   = O_W2   + SZ_SQW;
constexpr size_t O_WK   = O_WQ   + SZ_SQW;
constexpr size_t O_WV   = O_WK   + SZ_SQW;
constexpr size_t O_W3   = O_WV   + SZ_SQW;
constexpr size_t O_HENC = O_W3   + SZ_W3;               // 2 x bf16 state (double buffer)
constexpr size_t O_CENC = O_HENC + 2 * SZ_STATE * 2;    // fp32 state
constexpr size_t O_DH   = O_CENC + SZ_STATE * 4;        // 2 x bf16 state
constexpr size_t O_DC   = O_DH   + 2 * SZ_STATE * 2;    // fp32 state
constexpr size_t O_TOK  = O_DC   + SZ_STATE * 4;        // bf16 [16][64]
constexpr size_t O_Q    = O_TOK  + (size_t)Bsz * D0sz * 2;   // bf16 [16][512]
constexpr size_t O_ATT  = O_Q    + (size_t)Bsz * Hsz * 2;    // fp32 [16][1024]
constexpr size_t O_MAX  = O_ATT  + (size_t)Bsz * Lsz * 4;    // u32 [16]
constexpr size_t O_SUM  = O_MAX  + 64;                       // f32 [16]
constexpr size_t O_CTX  = O_SUM  + 64;                       // f32 [16][512]
constexpr size_t O_BAR  = O_CTX  + SZ_STATE * 4;             // barrier cnt/gen
constexpr size_t O_HS   = O_BAR  + 256;                      // bf16 [B][L][HE]
constexpr size_t O_HIDR = O_HS   + SZ_SEQ;
constexpr size_t O_KP   = O_HIDR + SZ_SEQ;
constexpr size_t O_VP   = O_KP   + SZ_SEQ;

// ---------------------------------------------------------------------------
// Types / helpers
// ---------------------------------------------------------------------------
typedef __bf16    v16bf __attribute__((ext_vector_type(16)));
typedef float     v8f   __attribute__((ext_vector_type(8)));
typedef unsigned  uint4v __attribute__((ext_vector_type(4)));

union Frag { uint4v u[2]; v16bf v; };

__device__ __forceinline__ v8f wmma_bf16(const Frag& a, const Frag& b, v8f c) {
  return __builtin_amdgcn_wmma_f32_16x16x32_bf16(false, a.v, false, b.v,
                                                 (short)0, c, false, false);
}

__device__ __forceinline__ unsigned short f2bf(float f) {
  unsigned u = __float_as_uint(f);
  return (unsigned short)((u + 0x7FFFu + ((u >> 16) & 1u)) >> 16);
}
__device__ __forceinline__ float b2f(unsigned short h) {
  return __uint_as_float(((unsigned)h) << 16);
}
__device__ __forceinline__ float sigm(float x) {
  return 1.0f / (1.0f + __expf(-x));
}
__device__ __forceinline__ unsigned flipf(float f) {
  unsigned u = __float_as_uint(f);
  return (u & 0x80000000u) ? ~u : (u | 0x80000000u);
}
__device__ __forceinline__ float unflipf(unsigned u) {
  return __uint_as_float((u & 0x80000000u) ? (u ^ 0x80000000u) : ~u);
}

// Software grid barrier across NWG co-resident workgroups (graph-safe;
// counters are re-initialized by pack_init_kernel on every launch).
__device__ __forceinline__ void grid_bar(unsigned* cnt, unsigned* gen, unsigned G) {
  __threadfence();
  __syncthreads();
  if (threadIdx.x == 0) {
    unsigned g = __atomic_load_n(gen, __ATOMIC_RELAXED);
    if (__atomic_fetch_add(cnt, 1u, __ATOMIC_ACQ_REL) == G - 1u) {
      __atomic_store_n(cnt, 0u, __ATOMIC_RELAXED);
      __atomic_fetch_add(gen, 1u, __ATOMIC_RELEASE);
    } else {
      while (__atomic_load_n(gen, __ATOMIC_ACQUIRE) == g)
        __builtin_amdgcn_s_sleep(2);
    }
  }
  __syncthreads();
  __threadfence();
}

// ---------------------------------------------------------------------------
// Kernel 0: convert weights fp32 -> bf16 packed layouts, zero recurrent state
// ---------------------------------------------------------------------------
__global__ void pack_init_kernel(const float* eWih, const float* eWhh,
                                 const float* dWih, const float* dWhh,
                                 const float* W2,  const float* Wq,
                                 const float* Wk,  const float* Wv,
                                 const float* W3,  char* ws) {
  unsigned short* encW = (unsigned short*)(ws + O_ENCW);
  unsigned short* decW = (unsigned short*)(ws + O_DECW);
  unsigned short* W2b  = (unsigned short*)(ws + O_W2);
  unsigned short* Wqb  = (unsigned short*)(ws + O_WQ);
  unsigned short* Wkb  = (unsigned short*)(ws + O_WK);
  unsigned short* Wvb  = (unsigned short*)(ws + O_WV);
  unsigned short* W3b  = (unsigned short*)(ws + O_W3);
  unsigned short* henc = (unsigned short*)(ws + O_HENC);
  float*          cenc = (float*)(ws + O_CENC);
  unsigned short* dhb  = (unsigned short*)(ws + O_DH);
  float*          dcb  = (float*)(ws + O_DC);
  float*          ctx  = (float*)(ws + O_CTX);
  unsigned*       maxb = (unsigned*)(ws + O_MAX);
  float*          sumb = (float*)(ws + O_SUM);
  unsigned*       bar  = (unsigned*)(ws + O_BAR);

  size_t i0 = (size_t)blockIdx.x * blockDim.x + threadIdx.x;
  size_t stride = (size_t)gridDim.x * blockDim.x;

  for (size_t i = i0; i < (size_t)G4 * KK; i += stride) {
    size_t n = i / KK, k = i % KK;
    encW[i] = f2bf(k < 64 ? eWih[n * 64 + k] : eWhh[n * 512 + (k - 64)]);
    decW[i] = f2bf(k < 64 ? dWih[n * 64 + k] : dWhh[n * 512 + (k - 64)]);
  }
  for (size_t i = i0; i < (size_t)Hsz * Hsz; i += stride) {
    W2b[i] = f2bf(W2[i]); Wqb[i] = f2bf(Wq[i]);
    Wkb[i] = f2bf(Wk[i]); Wvb[i] = f2bf(Wv[i]);
  }
  for (size_t i = i0; i < (size_t)D0sz * Hsz; i += stride) W3b[i] = f2bf(W3[i]);
  for (size_t i = i0; i < 2 * SZ_STATE; i += stride) { henc[i] = 0; dhb[i] = 0; }
  for (size_t i = i0; i < SZ_STATE; i += stride) { cenc[i] = 0.f; dcb[i] = 0.f; ctx[i] = 0.f; }
  if (i0 < 16) { maxb[i0] = 0u; sumb[i0] = 0.f; }
  if (i0 == 0) { bar[0] = 0u; bar[1] = 0u; }
}

// ---------------------------------------------------------------------------
// Kernel 1: encoder LSTM, 16 persistent WGs, grid barrier per timestep.
// Gate GEMM [16 x 2048, K=576] via v_wmma_f32_16x16x32_bf16.
// ---------------------------------------------------------------------------
__global__ void __launch_bounds__(256) encoder_kernel(const float* inputs,
                                                      const int* lens,
                                                      const float* enc_b,
                                                      char* ws) {
  unsigned short* encW = (unsigned short*)(ws + O_ENCW);
  unsigned short* henc = (unsigned short*)(ws + O_HENC);
  float*          cenc = (float*)(ws + O_CENC);
  unsigned short* hs   = (unsigned short*)(ws + O_HS);
  unsigned*       bar  = (unsigned*)(ws + O_BAR);

  __shared__ __align__(16) unsigned short act[16][KK];
  __shared__ float zbuf[16][128];

  const int w = blockIdx.x, tid = threadIdx.x;
  const int lane = tid & 31, r = tid >> 5;
  const int m_l = lane & 15, half = lane >> 4;

  for (int t = 0; t < Lsz; ++t) {
    const int cur = t & 1;
    const unsigned short* hold = henc + cur * SZ_STATE;
    unsigned short*       hnew = henc + (cur ^ 1) * SZ_STATE;

    // Stage [x_t | h_{t-1}] (16 x 576 bf16) in LDS
    for (int idx = tid; idx < 16 * KK; idx += 256) {
      int m = idx / KK, k = idx - m * KK;
      act[m][k] = (k < 64) ? f2bf(inputs[((size_t)m * Lsz + t) * 64 + k])
                           : hold[m * 512 + (k - 64)];
    }
    __syncthreads();

    // Each wave: one 16x16 gate tile, K=576 -> 18 WMMAs
    {
      const int n0 = (r >> 1) * 512 + w * 32 + (r & 1) * 16;
      const unsigned short* wbase = encW + (size_t)(n0 + m_l) * KK + half * 16;
      v8f c = {0.f, 0.f, 0.f, 0.f, 0.f, 0.f, 0.f, 0.f};
      for (int kt = 0; kt < 18; ++kt) {
        const int k0 = kt * 32;
        Frag fa, fb;
        fa.u[0] = *(const uint4v*)&act[m_l][k0 + half * 16];
        fa.u[1] = *(const uint4v*)&act[m_l][k0 + half * 16 + 8];
        fb.u[0] = *(const uint4v*)(wbase + k0);
        fb.u[1] = *(const uint4v*)(wbase + k0 + 8);
        c = wmma_bf16(fa, fb, c);
      }
      const int cb = (r >> 1) * 32 + (r & 1) * 16 + m_l;
#pragma unroll
      for (int i = 0; i < 8; ++i) zbuf[i + 8 * half][cb] = c[i];
    }
    __syncthreads();

    // LSTM cell for this WG's 32 h-columns
    {
      const int b = tid & 15, colg = tid >> 4;
#pragma unroll
      for (int it = 0; it < 2; ++it) {
        const int jl = colg + 16 * it, j = w * 32 + jl;
        float zi = zbuf[b][jl]       + enc_b[j];
        float zf = zbuf[b][32 + jl]  + enc_b[512 + j];
        float zg = zbuf[b][64 + jl]  + enc_b[1024 + j];
        float zo = zbuf[b][96 + jl]  + enc_b[1536 + j];
        float co = cenc[b * 512 + j];
        float cn = sigm(zf) * co + sigm(zi) * tanhf(zg);
        cenc[b * 512 + j] = cn;
        unsigned short hb = f2bf(sigm(zo) * tanhf(cn));
        hnew[b * 512 + j] = hb;
        hs[((size_t)b * Lsz + t) * 512 + j] = (t < lens[b]) ? hb : (unsigned short)0;
      }
    }
    grid_bar(bar, bar + 1, NWG);
  }
}

// ---------------------------------------------------------------------------
// Kernel 2: big WMMA GEMM  Out[16384x512] = act(A[16384x512] @ W^T + b)
// (used for lin_2 + relu, and K / V projections)
// ---------------------------------------------------------------------------
__global__ void __launch_bounds__(256) gemm_bf16_kernel(const unsigned short* A,
                                                        const unsigned short* W,
                                                        const float* bias,
                                                        unsigned short* Out,
                                                        int relu) {
  const int lane = threadIdx.x & 31, r = threadIdx.x >> 5;
  const int wid = blockIdx.x * 8 + r;        // 32768 tiles total
  const int tm = wid >> 5, tn = wid & 31;
  const int m_l = lane & 15, half = lane >> 4;

  const unsigned short* arow = A + (size_t)(tm * 16 + m_l) * 512;
  const unsigned short* wrow = W + (size_t)(tn * 16 + m_l) * 512 + half * 16;
  __builtin_prefetch(arow + 512, 0, 0);

  v8f c = {0.f, 0.f, 0.f, 0.f, 0.f, 0.f, 0.f, 0.f};
  for (int kt = 0; kt < 16; ++kt) {
    const int k0 = kt * 32;
    Frag fa, fb;
    fa.u[0] = *(const uint4v*)(arow + k0 + half * 16);
    fa.u[1] = *(const uint4v*)(arow + k0 + half * 16 + 8);
    fb.u[0] = *(const uint4v*)(wrow + k0);
    fb.u[1] = *(const uint4v*)(wrow + k0 + 8);
    c = wmma_bf16(fa, fb, c);
  }
  const int col = tn * 16 + m_l;
  const float bv = bias[col];
#pragma unroll
  for (int i = 0; i < 8; ++i) {
    float v = c[i] + bv;
    if (relu) v = fmaxf(v, 0.f);
    Out[(size_t)(tm * 16 + i + 8 * half) * 512 + col] = f2bf(v);
  }
}

// ---------------------------------------------------------------------------
// Kernel 3: tok0 = relu(hid_r[:, L-2, :] @ W3^T + b3)   (tiny, one-shot)
// ---------------------------------------------------------------------------
__global__ void tok0_kernel(char* ws, const float* b3) {
  const unsigned short* hidr = (const unsigned short*)(ws + O_HIDR);
  const unsigned short* W3b  = (const unsigned short*)(ws + O_W3);
  unsigned short*       tokb = (unsigned short*)(ws + O_TOK);
  const int idx = blockIdx.x * 256 + threadIdx.x;       // 0..1023
  if (idx >= Bsz * D0sz) return;
  const int b = idx >> 6, n = idx & 63;
  const unsigned short* hr = hidr + ((size_t)b * Lsz + (Lsz - 2)) * 512;
  const unsigned short* wr = W3b + (size_t)n * 512;
  float s = 0.f;
  for (int k = 0; k < 512; ++k) s += b2f(hr[k]) * b2f(wr[k]);
  s += b3[n];
  tokb[b * 64 + n] = f2bf(fmaxf(s, 0.f));
}

// ---------------------------------------------------------------------------
// Kernel 4: decoder LSTM + attention, 16 persistent WGs, 7 grid barriers/step
// ---------------------------------------------------------------------------
__global__ void __launch_bounds__(256) decoder_kernel(const int* lens,
                                                      const float* dec_b,
                                                      const float* bq,
                                                      const float* b3,
                                                      char* ws, float* out) {
  unsigned short* decW = (unsigned short*)(ws + O_DECW);
  unsigned short* Wqb  = (unsigned short*)(ws + O_WQ);
  unsigned short* W3b  = (unsigned short*)(ws + O_W3);
  unsigned short* dh   = (unsigned short*)(ws + O_DH);
  float*          dc   = (float*)(ws + O_DC);
  unsigned short* tokb = (unsigned short*)(ws + O_TOK);
  unsigned short* qb   = (unsigned short*)(ws + O_Q);
  float*          attnb = (float*)(ws + O_ATT);
  unsigned*       maxb  = (unsigned*)(ws + O_MAX);
  float*          sumb  = (float*)(ws + O_SUM);
  float*          ctx   = (float*)(ws + O_CTX);
  unsigned*       bar   = (unsigned*)(ws + O_BAR);
  const unsigned short* kp = (const unsigned short*)(ws + O_KP);
  const unsigned short* vp = (const unsigned short*)(ws + O_VP);
  float* dec_out  = out;
  float* attn_out = out + (size_t)Bsz * Lsz * D0sz;

  __shared__ __align__(16) unsigned short act[16][KK];
  __shared__ float zbuf[16][128];
  __shared__ float psum[16][16];
  __shared__ float red[64][4];

  const int w = blockIdx.x, tid = threadIdx.x;
  const int lane = tid & 31, r = tid >> 5;
  const int m_l = lane & 15, half = lane >> 4;
  const float scale = 0.044194173824159216f;  // 1/sqrt(512)

  for (int t = 0; t < Lsz; ++t) {
    const int cur = t & 1;
    const unsigned short* dhold = dh + cur * SZ_STATE;
    unsigned short*       dhnew = dh + (cur ^ 1) * SZ_STATE;
    float evals[4];

    // ---- Phase A: gates + cell; also reset ctx slice / max / sum --------
    for (int idx = tid; idx < 16 * KK; idx += 256) {
      int m = idx / KK, k = idx - m * KK;
      act[m][k] = (k < 64) ? tokb[m * 64 + k] : dhold[m * 512 + (k - 64)];
    }
    __syncthreads();
    {
      const int n0 = (r >> 1) * 512 + w * 32 + (r & 1) * 16;
      const unsigned short* wbase = decW + (size_t)(n0 + m_l) * KK + half * 16;
      v8f c = {0.f, 0.f, 0.f, 0.f, 0.f, 0.f, 0.f, 0.f};
      for (int kt = 0; kt < 18; ++kt) {
        const int k0 = kt * 32;
        Frag fa, fb;
        fa.u[0] = *(const uint4v*)&act[m_l][k0 + half * 16];
        fa.u[1] = *(const uint4v*)&act[m_l][k0 + half * 16 + 8];
        fb.u[0] = *(const uint4v*)(wbase + k0);
        fb.u[1] = *(const uint4v*)(wbase + k0 + 8);
        c = wmma_bf16(fa, fb, c);
      }
      const int cb = (r >> 1) * 32 + (r & 1) * 16 + m_l;
#pragma unroll
      for (int i = 0; i < 8; ++i) zbuf[i + 8 * half][cb] = c[i];
    }
    __syncthreads();
    {
      const int b = tid & 15, colg = tid >> 4;
#pragma unroll
      for (int it = 0; it < 2; ++it) {
        const int jl = colg + 16 * it, j = w * 32 + jl;
        float zi = zbuf[b][jl]       + dec_b[j];
        float zf = zbuf[b][32 + jl]  + dec_b[512 + j];
        float zg = zbuf[b][64 + jl]  + dec_b[1024 + j];
        float zo = zbuf[b][96 + jl]  + dec_b[1536 + j];
        float co = dc[b * 512 + j];
        float cn = sigm(zf) * co + sigm(zi) * tanhf(zg);
        dc[b * 512 + j] = cn;
        dhnew[b * 512 + j] = f2bf(sigm(zo) * tanhf(cn));
      }
#pragma unroll
      for (int it = 0; it < 2; ++it) {
        const int idx = tid * 2 + it;  // 0..511 -> 16b x 32 cols
        ctx[(idx >> 5) * 512 + w * 32 + (idx & 31)] = 0.f;
      }
      if (w == 0 && tid < 16) { maxb[tid] = 0u; sumb[tid] = 0.f; }
    }
    grid_bar(bar, bar + 1, NWG);

    // ---- Phase B: q = h @ Wq^T + bq (this WG: 2 N-tiles, K=512) ----------
    if (r < 2) {
      const int n0 = w * 32 + r * 16;
      const unsigned short* arow  = dhnew + m_l * 512;
      const unsigned short* wbase = Wqb + (size_t)(n0 + m_l) * 512 + half * 16;
      v8f c = {0.f, 0.f, 0.f, 0.f, 0.f, 0.f, 0.f, 0.f};
      for (int kt = 0; kt < 16; ++kt) {
        const int k0 = kt * 32;
        Frag fa, fb;
        fa.u[0] = *(const uint4v*)(arow + k0 + half * 16);
        fa.u[1] = *(const uint4v*)(arow + k0 + half * 16 + 8);
        fb.u[0] = *(const uint4v*)(wbase + k0);
        fb.u[1] = *(const uint4v*)(wbase + k0 + 8);
        c = wmma_bf16(fa, fb, c);
      }
      const int col = n0 + m_l;
      const float bv = bq[col];
#pragma unroll
      for (int i = 0; i < 8; ++i)
        qb[(size_t)(i + 8 * half) * 512 + col] = f2bf(c[i] + bv);
    }
    grid_bar(bar, bar + 1, NWG);

    // ---- Phase C: scores (wave-parallel dots, shfl reduce) + running max -
    for (int b = 0; b < 16; ++b) {
      float mx = -3.0e38f;
      for (int li = 0; li < 8; ++li) {
        const int l = w * 64 + r * 8 + li;
        const unsigned short* qrow = qb + b * 512 + lane * 16;
        const unsigned short* krow = kp + ((size_t)b * Lsz + l) * 512 + lane * 16;
        float s = 0.f;
#pragma unroll
        for (int i = 0; i < 16; ++i) s += b2f(qrow[i]) * b2f(krow[i]);
#pragma unroll
        for (int off = 16; off; off >>= 1) s += __shfl_xor(s, off, 32);
        s *= scale;
        if (l >= lens[b]) s = -1e9f;
        if (lane == 0) attnb[b * Lsz + l] = s;
        mx = fmaxf(mx, s);
      }
      if (lane == 0) atomicMax(&maxb[b], flipf(mx));
    }
    grid_bar(bar, bar + 1, NWG);

    // ---- Phase D: exp + partial sums ------------------------------------
    {
      const int b = tid & 15, g = tid >> 4;
      const float m = unflipf(maxb[b]);
      float part = 0.f;
#pragma unroll
      for (int i = 0; i < 4; ++i) {
        const int l = w * 64 + g * 4 + i;
        const float e = __expf(attnb[b * Lsz + l] - m);
        evals[i] = e;
        part += e;
      }
      psum[g][b] = part;
    }
    __syncthreads();
    if (tid < 16) {
      float s2 = 0.f;
#pragma unroll
      for (int g2 = 0; g2 < 16; ++g2) s2 += psum[g2][tid];
      unsafeAtomicAdd(&sumb[tid], s2);
    }
    grid_bar(bar, bar + 1, NWG);

    // ---- Phase E: normalize, emit attention weights ----------------------
    {
      const int b = tid & 15, g = tid >> 4;
      const float inv = 1.f / sumb[b];
#pragma unroll
      for (int i = 0; i < 4; ++i) {
        const int l = w * 64 + g * 4 + i;
        const float a = evals[i] * inv;
        attnb[b * Lsz + l] = a;
        attn_out[((size_t)b * Lsz + t) * Lsz + l] = a;
      }
    }
    grid_bar(bar, bar + 1, NWG);

    // ---- Phase F: ctx partial accumulation over this WG's l-slice --------
    {
      const int b = tid & 15, hg = tid >> 4;
      const int h0 = hg * 32;
      float acc[32];
#pragma unroll
      for (int i = 0; i < 32; ++i) acc[i] = 0.f;
      for (int li = 0; li < 64; ++li) {
        const int l = w * 64 + li;
        const float a = attnb[b * Lsz + l];
        const unsigned short* vr = vp + ((size_t)b * Lsz + l) * 512 + h0;
#pragma unroll
        for (int i = 0; i < 32; ++i) acc[i] += a * b2f(vr[i]);
      }
      for (int i = 0; i < 32; ++i)
        unsafeAtomicAdd(&ctx[b * 512 + h0 + i], acc[i]);
    }
    grid_bar(bar, bar + 1, NWG);

    // ---- Phase G: tok = relu(ctx @ W3^T + b3), write dec_out -------------
    {
      const int o = tid >> 2, part = tid & 3;
      const int b = o >> 2, cl = o & 3;
      const int n = w * 4 + cl;
      const unsigned short* wr = W3b + (size_t)n * 512 + part * 128;
      const float* cr = ctx + b * 512 + part * 128;
      float s = 0.f;
      for (int k = 0; k < 128; ++k) s += cr[k] * b2f(wr[k]);
      red[o][part] = s;
    }
    __syncthreads();
    if (tid < 64) {
      const int b = tid >> 2, cl = tid & 3;
      const int n = w * 4 + cl;
      float s = red[tid][0] + red[tid][1] + red[tid][2] + red[tid][3] + b3[n];
      const float tf = fmaxf(s, 0.f);
      tokb[b * 64 + n] = f2bf(tf);
      dec_out[((size_t)b * Lsz + t) * 64 + n] = (t < lens[b]) ? tf : 0.f;
    }
    grid_bar(bar, bar + 1, NWG);
  }
}

// ---------------------------------------------------------------------------
// Host launcher
// ---------------------------------------------------------------------------
extern "C" void kernel_launch(void* const* d_in, const int* in_sizes, int n_in,
                              void* d_out, int out_size, void* d_ws, size_t ws_size,
                              hipStream_t stream) {
  (void)in_sizes; (void)n_in; (void)out_size; (void)ws_size;
  const float* inputs = (const float*)d_in[0];
  const int*   lens   = (const int*)d_in[1];
  // d_in[2] = in_mask (derived from lens; not needed)
  const float* eWih = (const float*)d_in[3];
  const float* eWhh = (const float*)d_in[4];
  const float* eb   = (const float*)d_in[5];
  const float* W2   = (const float*)d_in[6];
  const float* b2   = (const float*)d_in[7];
  const float* dWih = (const float*)d_in[8];
  const float* dWhh = (const float*)d_in[9];
  const float* db   = (const float*)d_in[10];
  const float* Wq   = (const float*)d_in[11];
  const float* bq   = (const float*)d_in[12];
  const float* Wk   = (const float*)d_in[13];
  const float* bk   = (const float*)d_in[14];
  const float* Wv   = (const float*)d_in[15];
  const float* bv   = (const float*)d_in[16];
  const float* W3   = (const float*)d_in[17];
  const float* b3   = (const float*)d_in[18];

  char* ws = (char*)d_ws;
  unsigned short* hsb   = (unsigned short*)(ws + O_HS);
  unsigned short* hidr  = (unsigned short*)(ws + O_HIDR);
  unsigned short* kpb   = (unsigned short*)(ws + O_KP);
  unsigned short* vpb   = (unsigned short*)(ws + O_VP);
  unsigned short* W2b   = (unsigned short*)(ws + O_W2);
  unsigned short* Wkb   = (unsigned short*)(ws + O_WK);
  unsigned short* Wvb   = (unsigned short*)(ws + O_WV);

  // 0) weight packing + state init (also resets grid-barrier counters)
  pack_init_kernel<<<512, 256, 0, stream>>>(eWih, eWhh, dWih, dWhh,
                                            W2, Wq, Wk, Wv, W3, ws);
  // 1) encoder LSTM (persistent, grid barrier per step)
  encoder_kernel<<<NWG, 256, 0, stream>>>(inputs, lens, eb, ws);
  // 2) hid_r = relu(hs @ W2^T + b2)
  gemm_bf16_kernel<<<4096, 256, 0, stream>>>(hsb, W2b, b2, hidr, 1);
  // 3) kp / vp projections
  gemm_bf16_kernel<<<4096, 256, 0, stream>>>(hidr, Wkb, bk, kpb, 0);
  gemm_bf16_kernel<<<4096, 256, 0, stream>>>(hidr, Wvb, bv, vpb, 0);
  // 4) initial decoder token
  tok0_kernel<<<4, 256, 0, stream>>>(ws, b3);
  // 5) decoder LSTM + attention (persistent, grid barriers per step)
  decoder_kernel<<<NWG, 256, 0, stream>>>(lens, db, bq, b3, ws, (float*)d_out);
}
